// SharedExpertMoE_40355512714069
// MI455X (gfx1250) — matrix operand order repaired and checked by reading, not compiled, not run on hardware
//
#include <hip/hip_runtime.h>
#include <hip/hip_bf16.h>
#include <math.h>

typedef __bf16 bf16;
typedef __attribute__((ext_vector_type(16))) __bf16 v16bf;
typedef __attribute__((ext_vector_type(8)))  __bf16 v8bf;
typedef __attribute__((ext_vector_type(8)))  float  v8f;

#define NTOK   8192
#define DMODEL 1024
#define HSH    2048
#define HRT    1024
#define NEXP   16
#define TOPK   4

// A fragment: lane holds two contiguous 8-half chunks at p[0..7] and p[16..23]
__device__ __forceinline__ v16bf load_afrag(const bf16* p) {
    v8bf lo = *(const v8bf*)(p);
    v8bf hi = *(const v8bf*)(p + 16);
    return __builtin_shufflevector(lo, hi, 0,1,2,3,4,5,6,7,8,9,10,11,12,13,14,15);
}

// ---------------------------------------------------------------- utilities
__global__ void zero_f32_kernel(float* p, long n) {
    long i = (long)blockIdx.x * blockDim.x + threadIdx.x;
    if (i < n) p[i] = 0.0f;
}

__global__ void cvt_bf16_kernel(const float* __restrict__ in, bf16* __restrict__ out, long n) {
    long i = (long)blockIdx.x * blockDim.x + threadIdx.x;
    if (i < n) out[i] = (bf16)in[i];
}

// in: [E][K][N] f32 row-major  ->  out: [E][N][K] bf16 (K contiguous, for WMMA B frags)
__global__ void cvt_transpose_kernel(const float* __restrict__ in, bf16* __restrict__ out,
                                     int K, int N, long total) {
    long i = (long)blockIdx.x * blockDim.x + threadIdx.x;
    if (i >= total) return;
    long kn = (long)K * N;
    long e = i / kn;
    long r = i - e * kn;
    int  k = (int)(r / N);
    int  n = (int)(r - (long)k * N);
    out[e * kn + (long)n * K + k] = (bf16)in[i];
}

// ---------------------------------------------------------------- router
__global__ __launch_bounds__(256)
void router_kernel(const float* __restrict__ x, const float* __restrict__ Wr,
                   int* __restrict__ counts, int* __restrict__ tokens,
                   float* __restrict__ wts, float* __restrict__ load_sum) {
    int wave = threadIdx.x >> 5, lane = threadIdx.x & 31;
    int t = blockIdx.x * 8 + wave;
    if (t >= NTOK) return;

    float acc[NEXP];
#pragma unroll
    for (int e = 0; e < NEXP; e++) acc[e] = 0.0f;

    const float* xr = x + (long)t * DMODEL;
    for (int k = lane; k < DMODEL; k += 32) {
        float xv = xr[k];
        const float* wr = Wr + (long)k * NEXP;
#pragma unroll
        for (int e = 0; e < NEXP; e++) acc[e] += xv * wr[e];
    }
#pragma unroll
    for (int e = 0; e < NEXP; e++)
        for (int off = 16; off > 0; off >>= 1)
            acc[e] += __shfl_xor(acc[e], off, 32);

    if (lane == 0) {
        float mx = acc[0];
#pragma unroll
        for (int e = 1; e < NEXP; e++) mx = fmaxf(mx, acc[e]);
        float p[NEXP], den = 0.0f;
#pragma unroll
        for (int e = 0; e < NEXP; e++) { p[e] = expf(acc[e] - mx); den += p[e]; }
        float inv = 1.0f / den;
#pragma unroll
        for (int e = 0; e < NEXP; e++) atomicAdd(&load_sum[e], p[e] * inv);

        float lg[NEXP];
#pragma unroll
        for (int e = 0; e < NEXP; e++) lg[e] = acc[e];
        int   idx[TOPK];
        float val[TOPK];
#pragma unroll
        for (int j = 0; j < TOPK; j++) {
            int bi = 0; float bv = lg[0];
#pragma unroll
            for (int e = 1; e < NEXP; e++) { if (lg[e] > bv) { bv = lg[e]; bi = e; } }
            idx[j] = bi; val[j] = bv; lg[bi] = -3.0e38f;
        }
        float m2 = val[0], s2 = 0.0f, ww[TOPK];
#pragma unroll
        for (int j = 0; j < TOPK; j++) { ww[j] = expf(val[j] - m2); s2 += ww[j]; }
        float inv2 = 1.0f / s2;
#pragma unroll
        for (int j = 0; j < TOPK; j++) {
            int e = idx[j];
            int pos = atomicAdd(&counts[e], 1);
            tokens[e * NTOK + pos] = t;
            wts[e * NTOK + pos]    = ww[j] * inv2;
        }
    }
}

__global__ void balance_kernel(const float* __restrict__ load_sum, float* __restrict__ out) {
    if (threadIdx.x == 0 && blockIdx.x == 0) {
        float s = 0.0f;
#pragma unroll
        for (int e = 0; e < NEXP; e++) {
            float l = load_sum[e] * (1.0f / (float)NTOK);
            s += l * l;
        }
        out[0] = (float)NEXP * s;
    }
}

// ---------------------------------------------------------------- GEMM stage 1
// Block tile: 64(M) x 256(N); 8 waves (2M x 4N); wave tile 32(M) x 64(N):
// 2 A fragments, 4 B fragments, 8 WMMAs per K-step (each B feeds 2 WMMAs).
// H[m, n] = gelu( sum_k X[tok(m), k] * W1t[n, k] ),  K = DMODEL
__global__ __launch_bounds__(256)
void ffn_stage1(const bf16* __restrict__ Xb, const bf16* __restrict__ W1t,
                bf16* __restrict__ H, const int* __restrict__ tokens,
                const int* __restrict__ countPtr, int Mtotal, int Nsize) {
    int M = countPtr ? *countPtr : Mtotal;
    int m_base = blockIdx.y * 64;
    if (m_base >= M) return;

    int wave = threadIdx.x >> 5, lane = threadIdx.x & 31;
    int wm_base = m_base + (wave >> 2) * 32;
    int n_base  = blockIdx.x * 256 + (wave & 3) * 64;
    int g = lane >> 4, rm = lane & 15;

    int arow0 = wm_base + rm;      if (arow0 > M - 1) arow0 = M - 1;
    int arow1 = wm_base + 16 + rm; if (arow1 > M - 1) arow1 = M - 1;
    int tok0 = tokens ? tokens[arow0] : arow0;
    int tok1 = tokens ? tokens[arow1] : arow1;
    const bf16* ap0 = Xb + (size_t)tok0 * DMODEL + 8 * g;
    const bf16* ap1 = Xb + (size_t)tok1 * DMODEL + 8 * g;
    const bf16* bp  = W1t + (size_t)(n_base + rm) * DMODEL + 8 * g;
    const size_t BS = (size_t)16 * DMODEL;   // B column-block stride (elements)

    v8f c[8] = {};
#pragma unroll 2
    for (int kk = 0; kk < DMODEL; kk += 32) {
        v16bf a0 = load_afrag(ap0 + kk);
        v16bf a1 = load_afrag(ap1 + kk);
#pragma unroll
        for (int j = 0; j < 4; j++) {
            v16bf b = load_afrag(bp + (size_t)j * BS + kk);
            c[j]     = __builtin_amdgcn_wmma_f32_16x16x32_bf16(false, a0, false, b, (short)0, c[j],     false, false);
            c[4 + j] = __builtin_amdgcn_wmma_f32_16x16x32_bf16(false, a1, false, b, (short)0, c[4 + j], false, false);
        }
    }
#pragma unroll
    for (int mi = 0; mi < 2; mi++)
#pragma unroll
        for (int j = 0; j < 4; j++)
#pragma unroll
            for (int v = 0; v < 8; v++) {
                int gm = wm_base + mi * 16 + v + 8 * g;
                if (gm < M) {
                    float h = c[mi * 4 + j][v];
                    h = 0.5f * h * (1.0f + erff(h * 0.70710678118f));   // exact gelu
                    H[(size_t)gm * Nsize + n_base + j * 16 + rm] = (bf16)h;
                }
            }
}

// ---------------------------------------------------------------- GEMM stage 2
// Out[tok(m), n] (+)= w(m) * sum_k H[m, k] * W2t[n, k],  N = DMODEL
__global__ __launch_bounds__(256)
void ffn_stage2(const bf16* __restrict__ Hin, const bf16* __restrict__ W2t,
                float* __restrict__ Out, const int* __restrict__ tokens,
                const float* __restrict__ wts, const int* __restrict__ countPtr,
                int Mtotal, int Ksize, int accumulate) {
    int M = countPtr ? *countPtr : Mtotal;
    int m_base = blockIdx.y * 64;
    if (m_base >= M) return;

    int wave = threadIdx.x >> 5, lane = threadIdx.x & 31;
    int wm_base = m_base + (wave >> 2) * 32;
    int n_base  = blockIdx.x * 256 + (wave & 3) * 64;
    int g = lane >> 4, rm = lane & 15;

    int arow0 = wm_base + rm;      if (arow0 > M - 1) arow0 = M - 1;
    int arow1 = wm_base + 16 + rm; if (arow1 > M - 1) arow1 = M - 1;
    const bf16* ap0 = Hin + (size_t)arow0 * Ksize + 8 * g;
    const bf16* ap1 = Hin + (size_t)arow1 * Ksize + 8 * g;
    const bf16* bp  = W2t + (size_t)(n_base + rm) * Ksize + 8 * g;
    const size_t BS = (size_t)16 * Ksize;

    v8f c[8] = {};
#pragma unroll 2
    for (int kk = 0; kk < Ksize; kk += 32) {
        v16bf a0 = load_afrag(ap0 + kk);
        v16bf a1 = load_afrag(ap1 + kk);
#pragma unroll
        for (int j = 0; j < 4; j++) {
            v16bf b = load_afrag(bp + (size_t)j * BS + kk);
            c[j]     = __builtin_amdgcn_wmma_f32_16x16x32_bf16(false, a0, false, b, (short)0, c[j],     false, false);
            c[4 + j] = __builtin_amdgcn_wmma_f32_16x16x32_bf16(false, a1, false, b, (short)0, c[4 + j], false, false);
        }
    }
#pragma unroll
    for (int mi = 0; mi < 2; mi++)
#pragma unroll
        for (int j = 0; j < 4; j++)
#pragma unroll
            for (int v = 0; v < 8; v++) {
                int gm = wm_base + mi * 16 + v + 8 * g;
                if (gm < M) {
                    int tok  = tokens ? tokens[gm] : gm;
                    float w  = wts ? wts[gm] : 1.0f;
                    float vl = w * c[mi * 4 + j][v];
                    float* dst = &Out[(size_t)tok * DMODEL + n_base + j * 16 + rm];
                    if (accumulate) atomicAdd(dst, vl); else *dst = vl;
                }
            }
}

// ---------------------------------------------------------------- host
extern "C" void kernel_launch(void* const* d_in, const int* in_sizes, int n_in,
                              void* d_out, int out_size, void* d_ws, size_t ws_size,
                              hipStream_t stream) {
    const float* x   = (const float*)d_in[0];
    const float* W1s = (const float*)d_in[1];
    const float* W2s = (const float*)d_in[2];
    const float* Wr  = (const float*)d_in[3];
    const float* W1r = (const float*)d_in[4];
    const float* W2r = (const float*)d_in[5];
    float* out = (float*)d_out;

    char* ws = (char*)d_ws;
    size_t off = 0;
    bf16* Xb    = (bf16*)(ws + off); off += (size_t)NTOK * DMODEL * 2;
    bf16* W1sT  = (bf16*)(ws + off); off += (size_t)2 * HSH * DMODEL * 2;   // [e][N=2048][K=1024]
    bf16* W2sT  = (bf16*)(ws + off); off += (size_t)2 * DMODEL * HSH * 2;   // [e][N=1024][K=2048]
    bf16* W1rT  = (bf16*)(ws + off); off += (size_t)NEXP * HRT * DMODEL * 2;
    bf16* W2rT  = (bf16*)(ws + off); off += (size_t)NEXP * DMODEL * HRT * 2;
    bf16* Hbuf  = (bf16*)(ws + off); off += (size_t)NTOK * HSH * 2;         // reused by all experts
    int*   counts   = (int*)(ws + off);   off += NEXP * 4;
    float* load_sum = (float*)(ws + off); off += NEXP * 4;
    int*   tokens   = (int*)(ws + off);   off += (size_t)NEXP * NTOK * 4;
    float* wts      = (float*)(ws + off); off += (size_t)NEXP * NTOK * 4;

    const long OUTN = (long)NTOK * DMODEL + 1;
    zero_f32_kernel<<<(unsigned)((OUTN + 255) / 256), 256, 0, stream>>>(out, OUTN);
    zero_f32_kernel<<<1, 32, 0, stream>>>((float*)counts, 32); // counts + load_sum

    {
        long n = (long)NTOK * DMODEL;
        cvt_bf16_kernel<<<(unsigned)((n + 255) / 256), 256, 0, stream>>>(x, Xb, n);
        long t1 = (long)2 * DMODEL * HSH;
        cvt_transpose_kernel<<<(unsigned)((t1 + 255) / 256), 256, 0, stream>>>(W1s, W1sT, DMODEL, HSH, t1);
        cvt_transpose_kernel<<<(unsigned)((t1 + 255) / 256), 256, 0, stream>>>(W2s, W2sT, HSH, DMODEL, t1);
        long t2 = (long)NEXP * DMODEL * HRT;
        cvt_transpose_kernel<<<(unsigned)((t2 + 255) / 256), 256, 0, stream>>>(W1r, W1rT, DMODEL, HRT, t2);
        cvt_transpose_kernel<<<(unsigned)((t2 + 255) / 256), 256, 0, stream>>>(W2r, W2rT, HRT, DMODEL, t2);
    }

    router_kernel<<<NTOK / 8, 256, 0, stream>>>(x, Wr, counts, tokens, wts, load_sum);
    balance_kernel<<<1, 32, 0, stream>>>(load_sum, out + (size_t)NTOK * DMODEL);

    // shared experts: dense over all tokens
    for (int e = 0; e < 2; e++) {
        ffn_stage1<<<dim3(HSH / 256, NTOK / 64), 256, 0, stream>>>(
            Xb, W1sT + (size_t)e * HSH * DMODEL, Hbuf, nullptr, nullptr, NTOK, HSH);
        ffn_stage2<<<dim3(DMODEL / 256, NTOK / 64), 256, 0, stream>>>(
            Hbuf, W2sT + (size_t)e * DMODEL * HSH, out, nullptr, nullptr, nullptr,
            NTOK, HSH, e > 0 ? 1 : 0);
    }
    // routed experts: gathered sparse GEMMs, worst-case grid, early-exit on count
    for (int e = 0; e < NEXP; e++) {
        ffn_stage1<<<dim3(HRT / 256, NTOK / 64), 256, 0, stream>>>(
            Xb, W1rT + (size_t)e * HRT * DMODEL, Hbuf, tokens + (size_t)e * NTOK,
            counts + e, NTOK, HRT);
        ffn_stage2<<<dim3(DMODEL / 256, NTOK / 64), 256, 0, stream>>>(
            Hbuf, W2rT + (size_t)e * DMODEL * HRT, out, tokens + (size_t)e * NTOK,
            wts + (size_t)e * NTOK, counts + e, NTOK, HRT, 1);
    }
}
